// GCNNClassifierV5_69389491634796
// MI455X (gfx1250) — matrix operand order repaired and checked by reading, not compiled, not run on hardware
//
#include <hip/hip_runtime.h>
#include <hip/hip_bf16.h>
#include <math.h>

#define HEADS 4

typedef __attribute__((ext_vector_type(16))) __bf16 v16bf;
typedef __attribute__((ext_vector_type(8)))  float  v8f;

// ---------- helpers ----------
__device__ __forceinline__ unsigned f32_enc(float f) {
    unsigned u = __float_as_uint(f);
    return (u & 0x80000000u) ? ~u : (u | 0x80000000u);
}
__device__ __forceinline__ float f32_dec(unsigned e) {
    unsigned u = (e & 0x80000000u) ? (e & 0x7fffffffu) : ~e;
    return __uint_as_float(u);
}
__device__ __forceinline__ float gelu_exact(float x) {
    return 0.5f * x * (1.0f + erff(x * 0.7071067811865475f));
}
__device__ __forceinline__ float lrelu02(float x) { return x > 0.0f ? x : 0.2f * x; }

// ---------- generic fill ----------
__global__ void fill_u32_kernel(unsigned* __restrict__ p, unsigned v, long long n) {
    long long i = (long long)blockIdx.x * blockDim.x + threadIdx.x;
    if (i < n) p[i] = v;
}

// ---------- pack weights W[K,M] (f32) into bf16 WMMA-B lane layout ----------
// PW[((chunk*(M/16)+mt)*32 + lane)*16 + e] = bf16( W[(chunk*32 + 8*(lane>>4) + e + (e>=8?8:0)) * M
//                                                    + mt*16 + (lane&15)] ),  zero-filled K tail
__global__ void pack_w_bf16_kernel(const float* __restrict__ W, __bf16* __restrict__ PW,
                                   int K, int M, int total) {
    int t = blockIdx.x * blockDim.x + threadIdx.x;   // t = (chunk*mt16 + mt)*32 + lane
    if (t >= total) return;
    int lane = t & 31;
    int cm   = t >> 5;
    int mt16 = M / 16;
    int c  = cm / mt16;
    int mt = cm % mt16;
    int col = mt * 16 + (lane & 15);
    int kh  = lane >> 4;
    __bf16* out = PW + (size_t)t * 16;
#pragma unroll
    for (int e = 0; e < 16; ++e) {
        int k = c * 32 + 8 * kh + e + (e >= 8 ? 8 : 0);
        float v = (k < K) ? W[(size_t)k * M + col] : 0.0f;
        out[e] = (__bf16)v;
    }
}

// ---------- WMMA bf16 GEMM: Y[N,M] = X[N,K] @ W[K,M] (+bias) ----------
// One wave computes a 16x64 output strip: A converted once, 4 WMMAs per 32-K chunk.
// No branches in the K loop: A tail handled by clamped address + x{0,1} scale,
// B tail pre-zeroed by the pack kernel.
__global__ void gemm_wmma_bf16(const float* __restrict__ X, const __bf16* __restrict__ PW,
                               const float* __restrict__ bias, float* __restrict__ Y,
                               int K, int M, int tilesM4) {
    int tile = blockIdx.x;
    int tm4 = tile % tilesM4;
    int tn  = tile / tilesM4;
    int lane = threadIdx.x & 31;
    int r16  = lane & 15;
    int kh   = lane >> 4;
    int row  = tn * 16 + r16;
    const float* Xrow = X + (size_t)row * K;
    int chunks = (K + 31) / 32;
    int mt16   = M / 16;

    v8f acc[4] = {};
    for (int c = 0; c < chunks; ++c) {
        int kb = c * 32;
        // ---- A: 16x32 tile, this lane: K runs [k0,k0+7] and [k1,k1+7] (8-aligned) ----
        int k0 = kb + 8 * kh;
        int k1 = kb + 16 + 8 * kh;
        float s0 = (k0 + 8 <= K) ? 1.0f : 0.0f;
        float s1 = (k1 + 8 <= K) ? 1.0f : 0.0f;
        int kc0 = (k0 + 8 <= K) ? k0 : 0;      // clamped in-bounds address, value zeroed below
        int kc1 = (k1 + 8 <= K) ? k1 : 0;
        const float4* p0 = (const float4*)(Xrow + kc0);
        const float4* p1 = (const float4*)(Xrow + kc1);
        float4 x0 = p0[0], x1 = p0[1], x2 = p1[0], x3 = p1[1];
        v16bf a;
        a[0]  = (__bf16)(x0.x * s0); a[1]  = (__bf16)(x0.y * s0);
        a[2]  = (__bf16)(x0.z * s0); a[3]  = (__bf16)(x0.w * s0);
        a[4]  = (__bf16)(x1.x * s0); a[5]  = (__bf16)(x1.y * s0);
        a[6]  = (__bf16)(x1.z * s0); a[7]  = (__bf16)(x1.w * s0);
        a[8]  = (__bf16)(x2.x * s1); a[9]  = (__bf16)(x2.y * s1);
        a[10] = (__bf16)(x2.z * s1); a[11] = (__bf16)(x2.w * s1);
        a[12] = (__bf16)(x3.x * s1); a[13] = (__bf16)(x3.y * s1);
        a[14] = (__bf16)(x3.z * s1); a[15] = (__bf16)(x3.w * s1);
        // ---- B: 4 pre-packed 32x16 subtiles, 32B vector load per subtile ----
        const __bf16* bbase = PW + ((((size_t)c * mt16 + (size_t)tm4 * 4) * 32) + lane) * 16;
        v16bf b0 = *(const v16bf*)(bbase);
        v16bf b1 = *(const v16bf*)(bbase + 512);
        v16bf b2 = *(const v16bf*)(bbase + 1024);
        v16bf b3 = *(const v16bf*)(bbase + 1536);
        acc[0] = __builtin_amdgcn_wmma_f32_16x16x32_bf16(false, a, false, b0, (short)0, acc[0], false, false);
        acc[1] = __builtin_amdgcn_wmma_f32_16x16x32_bf16(false, a, false, b1, (short)0, acc[1], false, false);
        acc[2] = __builtin_amdgcn_wmma_f32_16x16x32_bf16(false, a, false, b2, (short)0, acc[2], false, false);
        acc[3] = __builtin_amdgcn_wmma_f32_16x16x32_bf16(false, a, false, b3, (short)0, acc[3], false, false);
    }
#pragma unroll
    for (int st = 0; st < 4; ++st) {
        int col = tm4 * 64 + st * 16 + r16;
        float bv = bias ? bias[col] : 0.0f;
#pragma unroll
        for (int r = 0; r < 8; ++r) {
            int orow = tn * 16 + r + 8 * kh;   // C/D: VGPR r -> row r / r+8 per lane half
            Y[(size_t)orow * M + col] = acc[st][r] + bv;
        }
    }
}

// ---------- attention coefficients a_src/a_dst per (node, head) ----------
__global__ void attn_coeff_kernel(const float* __restrict__ hW,
                                  const float* __restrict__ att_src,
                                  const float* __restrict__ att_dst,
                                  float* __restrict__ asrc, float* __restrict__ adst,
                                  int Nn, int O) {
    int t = blockIdx.x * blockDim.x + threadIdx.x;
    if (t >= Nn * HEADS) return;
    int n = t / HEADS, h = t % HEADS;
    int C = O / HEADS;
    const float* hp = hW + (size_t)n * O + h * C;
    const float* as = att_src + h * C;
    const float* ad = att_dst + h * C;
    float ss = 0.0f, sd = 0.0f;
    for (int c = 0; c < C; ++c) { float v = hp[c]; ss += v * as[c]; sd += v * ad[c]; }
    asrc[t] = ss;
    adst[t] = sd;
}

__device__ __forceinline__ void edge_sd(const int* __restrict__ ei, int e, int E,
                                        int* s, int* d) {
    if (e < E) { *s = ei[e]; *d = ei[E + e]; }
    else       { *s = *d = e - E; }   // self loops appended after real edges
}

// ---------- segment max over incoming edges ----------
__global__ void edge_max_kernel(const int* __restrict__ ei,
                                const float* __restrict__ asrc, const float* __restrict__ adst,
                                unsigned* __restrict__ menc, int E, int Etot) {
    int t = blockIdx.x * blockDim.x + threadIdx.x;
    if (t >= Etot * HEADS) return;
    int e = t / HEADS, h = t % HEADS, s, d;
    edge_sd(ei, e, E, &s, &d);
    float a = lrelu02(asrc[s * HEADS + h] + adst[d * HEADS + h]);
    atomicMax(&menc[d * HEADS + h], f32_enc(a));
}

// ---------- segment sum of exp(a - max) ----------
__global__ void edge_sum_kernel(const int* __restrict__ ei,
                                const float* __restrict__ asrc, const float* __restrict__ adst,
                                const unsigned* __restrict__ menc, float* __restrict__ ssum,
                                int E, int Etot) {
    int t = blockIdx.x * blockDim.x + threadIdx.x;
    if (t >= Etot * HEADS) return;
    int e = t / HEADS, h = t % HEADS, s, d;
    edge_sd(ei, e, E, &s, &d);
    float a = lrelu02(asrc[s * HEADS + h] + adst[d * HEADS + h]);
    float w = expf(a - f32_dec(menc[d * HEADS + h]));
    atomicAdd(&ssum[d * HEADS + h], w);
}

// ---------- weighted aggregation ----------
__global__ void edge_agg_kernel(const int* __restrict__ ei,
                                const float* __restrict__ asrc, const float* __restrict__ adst,
                                const unsigned* __restrict__ menc, const float* __restrict__ ssum,
                                const float* __restrict__ hW, float* __restrict__ agg,
                                int E, int Etot, int O) {
    int t = blockIdx.x * blockDim.x + threadIdx.x;
    if (t >= Etot * HEADS) return;
    int e = t / HEADS, h = t % HEADS, s, d;
    edge_sd(ei, e, E, &s, &d);
    int C = O / HEADS;
    float a = lrelu02(asrc[s * HEADS + h] + adst[d * HEADS + h]);
    float w = expf(a - f32_dec(menc[d * HEADS + h]));
    float alpha = w / (ssum[d * HEADS + h] + 1e-16f);
    const float* hp = hW + (size_t)s * O + h * C;
    float* op = agg + (size_t)d * O + h * C;
    for (int c = 0; c < C; ++c) atomicAdd(&op[c], hp[c] * alpha);
}

// ---------- fused (bias +) LayerNorm (+res) + GELU, one wave per row ----------
#define MAXR 16
__global__ void row_finalize_kernel(const float* __restrict__ in, const float* __restrict__ bias,
                                    const float* __restrict__ gam, const float* __restrict__ bet,
                                    const float* __restrict__ res, float* __restrict__ out,
                                    int Nn, int O, int do_ln) {
    int wid  = (blockIdx.x * blockDim.x + threadIdx.x) >> 5;
    int lane = threadIdx.x & 31;
    if (wid >= Nn) return;
    const float* ip = in + (size_t)wid * O;
    int nr = (O + 31) / 32;
    float v[MAXR];
    float sum = 0.0f, sumsq = 0.0f;
    for (int r = 0; r < nr; ++r) {
        int o = lane + 32 * r;
        float x = (o < O) ? (ip[o] + (bias ? bias[o] : 0.0f)) : 0.0f;
        v[r] = x;
        sum += x; sumsq += x * x;
    }
#pragma unroll
    for (int m = 16; m >= 1; m >>= 1) {
        sum   += __shfl_xor(sum, m, 32);
        sumsq += __shfl_xor(sumsq, m, 32);
    }
    float mean = sum / O;
    float var  = sumsq / O - mean * mean;
    float rstd = rsqrtf(var + 1e-5f);
    for (int r = 0; r < nr; ++r) {
        int o = lane + 32 * r;
        if (o >= O) continue;
        float x = v[r];
        float y = do_ln ? ((x - mean) * rstd * gam[o] + bet[o]) : x;
        if (res) y += res[(size_t)wid * O + o];
        out[(size_t)wid * O + o] = gelu_exact(y);
    }
}

// ---------- graph pooling (sum / max / count) ----------
__global__ void pool_kernel(const float* __restrict__ h, const int* __restrict__ batch,
                            float* __restrict__ gsum, unsigned* __restrict__ gmax,
                            float* __restrict__ gcnt, int Nn, int O) {
    int n = blockIdx.x * blockDim.x + threadIdx.x;
    if (n >= Nn) return;
    int b = batch[n];
    atomicAdd(&gcnt[b], 1.0f);
    const float* hp = h + (size_t)n * O;
    for (int o = 0; o < O; ++o) {
        float x = hp[o];
        atomicAdd(&gsum[(size_t)b * O + o], x);
        atomicMax(&gmax[(size_t)b * O + o], f32_enc(x));
    }
}

// ---------- gfeat = [mean, max, sum/10] -> concat buffer cols 0..383 ----------
__global__ void gfeat_kernel(const float* __restrict__ gsum, const unsigned* __restrict__ gmax,
                             const float* __restrict__ gcnt, float* __restrict__ cbuf,
                             int B, int O) {
    int t = blockIdx.x * blockDim.x + threadIdx.x;
    if (t >= B * O) return;
    int g = t / O, o = t % O;
    float s = gsum[t];
    float c = gcnt[g]; c = (c < 1.0f) ? 1.0f : c;
    cbuf[(size_t)g * 576 + o]         = s / c;
    cbuf[(size_t)g * 576 + O + o]     = f32_dec(gmax[t]);
    cbuf[(size_t)g * 576 + 2 * O + o] = s * 0.1f;
}

// ---------- copy global-branch features into concat buffer cols 384..575 ----------
__global__ void copy_g2_kernel(const float* __restrict__ src, float* __restrict__ cbuf, int B) {
    int t = blockIdx.x * blockDim.x + threadIdx.x;
    if (t >= B * 192) return;
    int g = t / 192, o = t % 192;
    cbuf[(size_t)g * 576 + 384 + o] = src[t];
}

// ---------- final classifier row (M=1, too small for a WMMA tile) ----------
__global__ void head_kernel(const float* __restrict__ z, const float* __restrict__ w,
                            const float* __restrict__ b, float* __restrict__ out, int B) {
    int g = blockIdx.x * blockDim.x + threadIdx.x;
    if (g >= B) return;
    float s = b[0];
    for (int i = 0; i < 64; ++i) s += z[(size_t)g * 64 + i] * w[i];
    out[g] = s;
}

// ================= host side =================
struct GatP { const float *W, *as, *ad, *bias, *g, *b, *rw, *rb; };

static inline void zero_buf(void* p, long long n_u32, hipStream_t s) {
    int bs = 256;
    long long gr = (n_u32 + bs - 1) / bs;
    fill_u32_kernel<<<dim3((unsigned)gr), dim3(bs), 0, s>>>((unsigned*)p, 0u, n_u32);
}

static inline void gemm(const float* X, const float* Wt, const float* bias, float* Y,
                        int Nr, int K, int M, __bf16* PW, hipStream_t s) {
    int chunks = (K + 31) / 32;
    int mt16 = M / 16;
    int total = chunks * mt16 * 32;
    pack_w_bf16_kernel<<<dim3((total + 255) / 256), dim3(256), 0, s>>>(Wt, PW, K, M, total);
    int tilesN = Nr / 16, tilesM4 = M / 64;
    gemm_wmma_bf16<<<dim3(tilesN * tilesM4), dim3(32), 0, s>>>(X, PW, bias, Y, K, M, tilesM4);
}

static inline void finalize(const float* in, const float* bias, const float* g, const float* b,
                            const float* res, float* out, int Nn, int O, int do_ln,
                            hipStream_t s) {
    int waves_per_block = 4, bs = 32 * waves_per_block;
    int grid = (Nn + waves_per_block - 1) / waves_per_block;
    row_finalize_kernel<<<dim3(grid), dim3(bs), 0, s>>>(in, bias, g, b, res, out, Nn, O, do_ln);
}

extern "C" void kernel_launch(void* const* d_in, const int* in_sizes, int n_in,
                              void* d_out, int out_size, void* d_ws, size_t ws_size,
                              hipStream_t stream) {
    const float* x   = (const float*)d_in[0];
    const int*   ei  = (const int*)d_in[1];
    const int*   bat = (const int*)d_in[2];
    const float* u   = (const float*)d_in[3];

    const int N = in_sizes[0] / 16;        // nodes
    const int E = in_sizes[1] / 2;         // edges
    const int B = in_sizes[3] / 64;        // graphs
    const int Etot = E + N;                // + self loops

    // ---- walk params (dict insertion order from the reference) ----
    int pi = 4;
    auto nx = [&]() { return (const float*)d_in[pi++]; };
    const float* emb_w = nx(); const float* emb_b = nx();
    const float* emb_g = nx(); const float* emb_bt = nx();
    GatP g1{}, g2{}, g3{}, g4{};
    auto read_gat = [&](GatP& gp, bool has_res) {
        gp.W = nx(); gp.as = nx(); gp.ad = nx(); gp.bias = nx();
        gp.g = nx(); gp.b = nx();
        if (has_res) { gp.rw = nx(); gp.rb = nx(); } else { gp.rw = nullptr; gp.rb = nullptr; }
    };
    read_gat(g1, true); read_gat(g2, true); read_gat(g3, false); read_gat(g4, true);
    const float* ge1_w = nx(); const float* ge1_b = nx();
    const float* ge1_g = nx(); const float* ge1_bt = nx();
    const float* ge2_w = nx(); const float* ge2_b = nx();
    const float* ge2_g = nx(); const float* ge2_bt = nx();
    const float* fu1_w = nx(); const float* fu1_b = nx();
    const float* fu1_g = nx(); const float* fu1_bt = nx();
    const float* fu2_w = nx(); const float* fu2_b = nx();
    const float* fu2_g = nx(); const float* fu2_bt = nx();
    const float* cl1_w = nx(); const float* cl1_b = nx();
    const float* cl2_w = nx(); const float* cl2_b = nx();

    // ---- workspace layout (floats) ----
    float* ws = (float*)d_ws;
    size_t off = 0;
    float* bufA = ws + off; off += (size_t)N * 256;
    float* bufB = ws + off; off += (size_t)N * 256;
    float* HW   = ws + off; off += (size_t)N * 256;
    float* AGG  = ws + off; off += (size_t)N * 256;
    float* RES  = ws + off; off += (size_t)N * 256;
    float*    ASRC = ws + off; off += (size_t)N * HEADS;
    float*    ADST = ws + off; off += (size_t)N * HEADS;
    unsigned* MENC = (unsigned*)(ws + off); off += (size_t)N * HEADS;
    float*    SSUM = ws + off; off += (size_t)N * HEADS;
    float*    GSUM  = ws + off; off += (size_t)B * 128;
    unsigned* GMAXE = (unsigned*)(ws + off); off += (size_t)B * 128;
    float*    GCNT  = ws + off; off += (size_t)B;
    float*    CBUF  = ws + off; off += (size_t)B * 576;
    float*    T1    = ws + off; off += (size_t)B * 384;
    float*    T2    = ws + off; off += (size_t)B * 384;
    float*    T3    = ws + off; off += (size_t)B * 192;
    float*    Z     = ws + off; off += (size_t)B * 64;
    __bf16*   PW    = (__bf16*)(ws + off); off += 131072;   // packed bf16 weights (<=443KB)

    const int EB = 256;
    const int egrid = (Etot * HEADS + EB - 1) / EB;

    // ---- embed: gelu(LN(x @ W + b)) ----
    gemm(x, emb_w, emb_b, HW, N, 16, 64, PW, stream);
    finalize(HW, nullptr, emb_g, emb_bt, nullptr, bufA, N, 64, 1, stream);

    // ---- one GAT block ----
    auto run_gat = [&](const float* Xin, float* Xout, int I, int O, const GatP& gp) {
        gemm(Xin, gp.W, nullptr, HW, N, I, O, PW, stream);   // h = X @ W  (WMMA)
        zero_buf(MENC, (long long)N * HEADS, stream);
        zero_buf(SSUM, (long long)N * HEADS, stream);
        zero_buf(AGG,  (long long)N * O,     stream);
        attn_coeff_kernel<<<dim3((N * HEADS + EB - 1) / EB), dim3(EB), 0, stream>>>(
            HW, gp.as, gp.ad, ASRC, ADST, N, O);
        edge_max_kernel<<<dim3(egrid), dim3(EB), 0, stream>>>(ei, ASRC, ADST, MENC, E, Etot);
        edge_sum_kernel<<<dim3(egrid), dim3(EB), 0, stream>>>(ei, ASRC, ADST, MENC, SSUM, E, Etot);
        edge_agg_kernel<<<dim3(egrid), dim3(EB), 0, stream>>>(ei, ASRC, ADST, MENC, SSUM, HW,
                                                              AGG, E, Etot, O);
        const float* resp;
        if (gp.rw) { gemm(Xin, gp.rw, gp.rb, RES, N, I, O, PW, stream); resp = RES; }
        else       { resp = Xin; }
        finalize(AGG, gp.bias, gp.g, gp.b, resp, Xout, N, O, 1, stream);
    };

    run_gat(bufA, bufB,  64, 128, g1);
    run_gat(bufB, bufA, 128, 256, g2);
    run_gat(bufA, bufB, 256, 256, g3);
    run_gat(bufB, bufA, 256, 128, g4);

    // ---- graph pooling -> CBUF[:, 0:384] ----
    zero_buf(GSUM,  (long long)B * 128, stream);
    zero_buf(GMAXE, (long long)B * 128, stream);
    zero_buf(GCNT,  (long long)B,       stream);
    pool_kernel<<<dim3((N + EB - 1) / EB), dim3(EB), 0, stream>>>(bufA, bat, GSUM, GMAXE, GCNT,
                                                                  N, 128);
    gfeat_kernel<<<dim3((B * 128 + EB - 1) / EB), dim3(EB), 0, stream>>>(GSUM, GMAXE, GCNT,
                                                                         CBUF, B, 128);

    // ---- global branch -> CBUF[:, 384:576] ----
    gemm(u, ge1_w, ge1_b, T1, B, 64, 256, PW, stream);
    finalize(T1, nullptr, ge1_g, ge1_bt, nullptr, T2, B, 256, 1, stream);
    gemm(T2, ge2_w, ge2_b, T1, B, 256, 192, PW, stream);
    finalize(T1, nullptr, ge2_g, ge2_bt, nullptr, T3, B, 192, 1, stream);
    copy_g2_kernel<<<dim3((B * 192 + EB - 1) / EB), dim3(EB), 0, stream>>>(T3, CBUF, B);

    // ---- fusion MLP ----
    gemm(CBUF, fu1_w, fu1_b, T1, B, 576, 384, PW, stream);
    finalize(T1, nullptr, fu1_g, fu1_bt, nullptr, T2, B, 384, 1, stream);
    gemm(T2, fu2_w, fu2_b, T1, B, 384, 192, PW, stream);
    finalize(T1, nullptr, fu2_g, fu2_bt, nullptr, T3, B, 192, 1, stream);

    // ---- classifier head ----
    gemm(T3, cl1_w, cl1_b, T1, B, 192, 64, PW, stream);
    finalize(T1, nullptr, nullptr, nullptr, nullptr, Z, B, 64, 0, stream);  // gelu only
    head_kernel<<<dim3((B + EB - 1) / EB), dim3(EB), 0, stream>>>(Z, cl2_w, cl2_b,
                                                                  (float*)d_out, B);
}
